// STDPAttention_57964878626849
// MI455X (gfx1250) — matrix operand order repaired and checked by reading, not compiled, not run on hardware
//
#include <hip/hip_runtime.h>
#include <hip/hip_bf16.h>

#define B_  2
#define NH_ 4
#define T_  1024
#define NK_ 8192
#define D_  256

typedef __bf16 bf16_t;
typedef __attribute__((ext_vector_type(16))) __bf16 bf16x16;
typedef __attribute__((ext_vector_type(8)))  __bf16 bf16x8;
typedef __attribute__((ext_vector_type(4)))  __bf16 bf16x4;
typedef __attribute__((ext_vector_type(8)))  float  f32x8;

union FragBF { bf16x16 v; uint4 q[2]; };
union Pack8  { bf16x8  v; uint4 q;    };

__device__ __forceinline__ float sigmoidf_(float x) { return 1.0f / (1.0f + __expf(-x)); }

// CDNA5 async DMA: per-lane 16B copy global -> LDS, tracked by ASYNCcnt.
__device__ __forceinline__ void async_load_b128(void* lds_ptr, const void* gptr) {
  uint32_t lds_off = (uint32_t)(uintptr_t)lds_ptr;   // flat->LDS: low 32 bits
  asm volatile("global_load_async_to_lds_b128 %0, %1, off"
               :
               : "v"(lds_off), "v"((unsigned long long)(uintptr_t)gptr)
               : "memory");
}
__device__ __forceinline__ void wait_asynccnt0() {
  asm volatile("s_wait_asynccnt 0x0" ::: "memory");
}

// ---------------------------------------------------------------------------
// Kernel 1: RoPE + fp32 -> bf16, QR written to workspace.
// One thread = 4 consecutive elements (2 rotation pairs).
// ---------------------------------------------------------------------------
__global__ void rope_kernel(const float* __restrict__ Q, bf16_t* __restrict__ qr) {
  const float INV2PI = 0.15915494309189535f;
  const float TWOPI  = 6.283185307179586f;
  size_t idx = ((size_t)blockIdx.x * blockDim.x + threadIdx.x) * 4;
  int n = (int)(idx & (NK_ - 1));            // even (idx multiple of 4)
  int t = (int)((idx >> 13) & (T_ - 1));     // N = 2^13, T = 2^10
  float4 v = *(const float4*)(Q + idx);
  // freqs = theta^(-q/N)/(2pi), theta = 2^16 -> exp2(-16*q/N)
  float f0 = exp2f(-16.0f * (float)n       / (float)NK_) * INV2PI;
  float f1 = exp2f(-16.0f * (float)(n + 2) / (float)NK_) * INV2PI;
  float p0 = (float)t * f0; p0 = (p0 - floorf(p0)) * TWOPI;
  float p1 = (float)t * f1; p1 = (p1 - floorf(p1)) * TWOPI;
  float s0, c0, s1, c1;
  __sincosf(p0, &s0, &c0);
  __sincosf(p1, &s1, &c1);
  bf16x4 o;
  o.x = (__bf16)(v.x * c0 - v.y * s0);
  o.y = (__bf16)(v.y * c0 + v.x * s0);
  o.z = (__bf16)(v.z * c1 - v.w * s1);
  o.w = (__bf16)(v.w * c1 + v.z * s1);
  *(bf16x4*)(qr + idx) = o;
}

// ---------------------------------------------------------------------------
// Kernel 2: gate[b,h,t] = 0.5 + 0.5*sigmoid(mean_D(traces))
// One 256-thread block per row (D = 256).
// ---------------------------------------------------------------------------
__global__ void gate_kernel(const float* __restrict__ traces, float* __restrict__ gate) {
  __shared__ float red[256];
  int row = blockIdx.x;
  int tid = threadIdx.x;
  red[tid] = traces[(size_t)row * D_ + tid];
  __syncthreads();
  for (int off = 128; off > 0; off >>= 1) {
    if (tid < off) red[tid] += red[tid + off];
    __syncthreads();
  }
  if (tid == 0) gate[row] = 0.5f + 0.5f * sigmoidf_(red[0] * (1.0f / (float)D_));
}

// ---------------------------------------------------------------------------
// Kernel 3: scores = tril(QR.QR^T * (1+STDP), -1) * gate   (bf16 WMMA, f32 acc)
// Block tile 128x128, 8 waves, wave tile 32x64 (2x4 accumulators).
// grid = (8 col tiles, 8 row tiles, 8 bh). Upper tiles zero-fill only.
// Double-buffered LDS; staging via global_load_async_to_lds_b128 (ASYNCcnt)
// overlapped with WMMA compute; one barrier per K-step.
// ---------------------------------------------------------------------------
__global__ __launch_bounds__(256)
void scores_kernel(const bf16_t* __restrict__ qr, const float* __restrict__ gate,
                   const float* __restrict__ stdp_scale, float* __restrict__ scores) {
  __shared__ bf16_t lA[2][128][72];   // [buf][t-row][k], +8 pad: conflict-free b128
  __shared__ bf16_t lB[2][128][72];   // [buf][s-row][k]
  const int tid = threadIdx.x;
  const int ct = blockIdx.x, rt = blockIdx.y, bh = blockIdx.z;
  const int t0 = rt * 128, s0 = ct * 128;
  float* scoresBH = scores + (size_t)bh * T_ * T_;

  if (ct > rt) {  // strictly upper tile: masked to zero
    for (int p = 0; p < 64; ++p) {
      int idx = p * 256 + tid;
      int r = idx >> 5, c = (idx & 31) * 4;
      *(float4*)(scoresBH + (size_t)(t0 + r) * T_ + s0 + c) = make_float4(0.f, 0.f, 0.f, 0.f);
    }
    return;
  }

  const bf16_t* qrBH = qr + (size_t)bh * T_ * NK_;
  const int lane = tid & 31, wave = tid >> 5;
  const int g = lane >> 4, lr = lane & 15;
  const int wr = wave >> 1, wc = wave & 1;

  f32x8 acc[2][4];
  for (int i = 0; i < 2; ++i)
    for (int j = 0; j < 4; ++j)
      for (int e = 0; e < 8; ++e) acc[i][j][e] = 0.0f;

  const int lrow = tid >> 3;        // 0..31
  const int lcol = (tid & 7) * 8;   // 0..56

  auto issue_tiles = [&](int buf, int kk) {
    for (int p = 0; p < 4; ++p) {
      int r = p * 32 + lrow;
      async_load_b128(&lA[buf][r][lcol], qrBH + (size_t)(t0 + r) * NK_ + kk + lcol);
      async_load_b128(&lB[buf][r][lcol], qrBH + (size_t)(s0 + r) * NK_ + kk + lcol);
    }
  };

  issue_tiles(0, 0);
  int buf = 0;
  for (int kk = 0; kk < NK_; kk += 64) {
    wait_asynccnt0();                // this wave's DMA for `buf` landed
    __syncthreads();                 // all waves' DMA landed; other buffer free
    if (kk + 64 < NK_) issue_tiles(buf ^ 1, kk + 64);  // DMA next, overlap compute
    for (int ks = 0; ks < 64; ks += 32) {
      FragBF fa[2], fb[4];
      for (int i = 0; i < 2; ++i) {          // A 16x32 bf16: K interleaved by 8
        int ar = wr * 32 + i * 16 + lr;
        fa[i].q[0] = *(const uint4*)&lA[buf][ar][ks + g * 8];
        fa[i].q[1] = *(const uint4*)&lA[buf][ar][ks + 16 + g * 8];
      }
      for (int j = 0; j < 4; ++j) {          // B 32x16 bf16: K interleaved by 8
        int br = wc * 64 + j * 16 + lr;
        fb[j].q[0] = *(const uint4*)&lB[buf][br][ks + g * 8];
        fb[j].q[1] = *(const uint4*)&lB[buf][br][ks + 16 + g * 8];
      }
      for (int i = 0; i < 2; ++i)
        for (int j = 0; j < 4; ++j)
          acc[i][j] = __builtin_amdgcn_wmma_f32_16x16x32_bf16(
              false, fa[i].v, false, fb[j].v, (short)0, acc[i][j], false, false);
    }
    buf ^= 1;
  }

  // Epilogue: STDP(LTP only survives strict tril) + mask + gate, write fp32.
  const float scale = stdp_scale[0];
  const float* gateBH = gate + bh * T_;
  for (int i = 0; i < 2; ++i)
    for (int j = 0; j < 4; ++j) {
      int srow = s0 + wc * 64 + j * 16 + lr;
      for (int e = 0; e < 8; ++e) {
        int trow = t0 + wr * 32 + i * 16 + g * 8 + e;
        float val = acc[i][j][e];
        if (trow > srow) {
          float dt = (float)(trow - srow);
          val *= (1.0f + scale * 0.01f * __expf(-dt * 0.05f)) * gateBH[trow];
        } else {
          val = 0.0f;
        }
        scoresBH[(size_t)trow * T_ + srow] = val;
      }
    }
}

// ---------------------------------------------------------------------------
// Kernel 4: output = scores @ V   (scores fp32 -> bf16 on LDS fill)
// Block tile 128(t) x 128(d), grid = (2, 8, 8). K-loop skips zero upper part.
// ---------------------------------------------------------------------------
__global__ __launch_bounds__(256)
void output_kernel(const float* __restrict__ scores, const float* __restrict__ V,
                   float* __restrict__ out) {
  __shared__ bf16_t lA[128][72];   // [t-row][k=s]
  __shared__ bf16_t lB[128][72];   // [d-col][k=s]  (V transposed into LDS)
  const int tid = threadIdx.x;
  const int ct = blockIdx.x, rt = blockIdx.y, bh = blockIdx.z;
  const int b = bh >> 2;
  const int t0 = rt * 128, d0 = ct * 128;
  const float* scoresBH = scores + (size_t)bh * T_ * T_;
  const float* Vb = V + (size_t)b * T_ * D_;
  const int lane = tid & 31, wave = tid >> 5;
  const int g = lane >> 4, lr = lane & 15;
  const int wr = wave >> 1, wc = wave & 1;

  f32x8 acc[2][4];
  for (int i = 0; i < 2; ++i)
    for (int j = 0; j < 4; ++j)
      for (int e = 0; e < 8; ++e) acc[i][j][e] = 0.0f;

  const int kEnd = (t0 + 128 < T_) ? (t0 + 128) : T_;  // scores[t][s]==0 for s>=t
  for (int kk = 0; kk < kEnd; kk += 64) {
    __syncthreads();
    // A tile: scores[t0..t0+127][kk..kk+63] fp32 -> bf16
    for (int p = 0; p < 4; ++p) {
      int r = p * 32 + (tid >> 3);
      int c = (tid & 7) * 8;
      const float* src = scoresBH + (size_t)(t0 + r) * T_ + kk + c;
      float4 u0 = *(const float4*)(src);
      float4 u1 = *(const float4*)(src + 4);
      Pack8 pk;
      pk.v[0] = (__bf16)u0.x; pk.v[1] = (__bf16)u0.y; pk.v[2] = (__bf16)u0.z; pk.v[3] = (__bf16)u0.w;
      pk.v[4] = (__bf16)u1.x; pk.v[5] = (__bf16)u1.y; pk.v[6] = (__bf16)u1.z; pk.v[7] = (__bf16)u1.w;
      *(uint4*)&lA[r][c] = pk.q;
    }
    // B tile: V[kk..kk+63][d0..d0+127] fp32 -> bf16, transposed to [d][k]
    for (int p = 0; p < 4; ++p) {
      int r = p * 16 + (tid >> 4);     // k within tile: 0..63
      int c = (tid & 15) * 8;          // d within tile: 0..120
      const float* src = Vb + (size_t)(kk + r) * D_ + d0 + c;
      float4 u0 = *(const float4*)(src);
      float4 u1 = *(const float4*)(src + 4);
      lB[c + 0][r] = (__bf16)u0.x; lB[c + 1][r] = (__bf16)u0.y;
      lB[c + 2][r] = (__bf16)u0.z; lB[c + 3][r] = (__bf16)u0.w;
      lB[c + 4][r] = (__bf16)u1.x; lB[c + 5][r] = (__bf16)u1.y;
      lB[c + 6][r] = (__bf16)u1.z; lB[c + 7][r] = (__bf16)u1.w;
    }
    __syncthreads();
    for (int ks = 0; ks < 64; ks += 32) {
      FragBF fa[2], fb[4];
      for (int i = 0; i < 2; ++i) {
        int ar = wr * 32 + i * 16 + lr;
        fa[i].q[0] = *(const uint4*)&lA[ar][ks + g * 8];
        fa[i].q[1] = *(const uint4*)&lA[ar][ks + 16 + g * 8];
      }
      for (int j = 0; j < 4; ++j) {
        int bc = wc * 64 + j * 16 + lr;
        fb[j].q[0] = *(const uint4*)&lB[bc][ks + g * 8];
        fb[j].q[1] = *(const uint4*)&lB[bc][ks + 16 + g * 8];
      }
      for (int i = 0; i < 2; ++i)
        for (int j = 0; j < 4; ++j)
          acc[i][j] = __builtin_amdgcn_wmma_f32_16x16x32_bf16(
              false, fa[i].v, false, fb[j].v, (short)0, acc[i][j], false, false);
    }
  }

  for (int i = 0; i < 2; ++i)
    for (int j = 0; j < 4; ++j) {
      int dcol = d0 + wc * 64 + j * 16 + lr;
      for (int e = 0; e < 8; ++e) {
        int trow = t0 + wr * 32 + i * 16 + g * 8 + e;
        out[((size_t)bh * T_ + trow) * D_ + dcol] = acc[i][j][e];
      }
    }
}

// ---------------------------------------------------------------------------
extern "C" void kernel_launch(void* const* d_in, const int* in_sizes, int n_in,
                              void* d_out, int out_size, void* d_ws, size_t ws_size,
                              hipStream_t stream) {
  (void)in_sizes; (void)n_in; (void)out_size; (void)ws_size;
  const float* Q          = (const float*)d_in[0];   // [B,nh,T,N]
  const float* V          = (const float*)d_in[1];   // [B,1,T,D]
  const float* traces     = (const float*)d_in[2];   // [B,nh,T,D]
  const float* stdp_scale = (const float*)d_in[3];   // [1]

  float* out    = (float*)d_out;                        // [B,nh,T,D]
  float* scores = out + (size_t)B_ * NH_ * T_ * D_;     // [B,nh,T,T]

  bf16_t* qr   = (bf16_t*)d_ws;                                          // 128 MB
  float*  gate = (float*)((char*)d_ws + (size_t)B_ * NH_ * T_ * NK_ * sizeof(bf16_t));

  const int ropeBlocks = (B_ * NH_ * T_ * NK_ / 4) / 256;  // 65536
  rope_kernel<<<ropeBlocks, 256, 0, stream>>>(Q, qr);
  gate_kernel<<<B_ * NH_ * T_, 256, 0, stream>>>(traces, gate);
  scores_kernel<<<dim3(8, 8, 8), 256, 0, stream>>>(qr, gate, stdp_scale, scores);
  output_kernel<<<dim3(2, 8, 8), 256, 0, stream>>>(scores, V, out);
}
